// MultiHeadAttention_50096498541218
// MI455X (gfx1250) — compile-verified, matrix operand order checked by reading, not run on hardware
//
#include <hip/hip_runtime.h>
#include <hip/hip_bf16.h>
#include <math.h>

#define D_MODEL 1024
#define N_HEADS 16
#define HEAD_DIM 64
#define SEQ 2048
#define BATCH 2
#define NTOK (BATCH * SEQ) /* 4096 tokens */

typedef __attribute__((ext_vector_type(4)))  __bf16 v4bf;
typedef __attribute__((ext_vector_type(8)))  __bf16 v8bf;
typedef __attribute__((ext_vector_type(16))) __bf16 v16bf;
typedef __attribute__((ext_vector_type(8)))  float  v8f;

static __device__ inline v8f vzero8() {
  v8f z = {0.f, 0.f, 0.f, 0.f, 0.f, 0.f, 0.f, 0.f};
  return z;
}

static __device__ inline v16bf cat16(v8bf a, v8bf b) {
  return __builtin_shufflevector(a, b, 0, 1, 2, 3, 4, 5, 6, 7,
                                 8, 9, 10, 11, 12, 13, 14, 15);
}

// Load a 16x32 bf16 WMMA A/B fragment from a row-major matrix with leading
// dimension `ld`. Per the CDNA5 ISA 16-bit A-matrix layout (16x32, wave32):
// lane holds row (lane&15); K chunks [8*half, 8*half+8) in VGPR0..3 and
// [16+8*half, 24+8*half) in VGPR4..7, where half = lane>>4.
static __device__ inline v16bf load_frag(const __bf16* base, int ld, int row,
                                         int k0, int half) {
  const __bf16* r = base + (size_t)row * ld + k0 + 8 * half;
  v8bf c0 = *reinterpret_cast<const v8bf*>(r);
  v8bf c1 = *reinterpret_cast<const v8bf*>(r + 16);
  return cat16(c0, c1);
}

static __device__ inline v8f wmma_bf16(v16bf a, v16bf b, v8f c) {
  // 8 args: (neg_a, A, neg_b, B, c_mod, C, reuse_a, reuse_b)
  return __builtin_amdgcn_wmma_f32_16x16x32_bf16(false, a, false, b,
                                                 (short)0, c, false, false);
}

// Reductions across the 16 lanes of a wave32 half (xor masks 1,2,4,8 stay
// within each 16-lane group).
static __device__ inline float redmax16(float v) {
  v = fmaxf(v, __shfl_xor(v, 1, 32));
  v = fmaxf(v, __shfl_xor(v, 2, 32));
  v = fmaxf(v, __shfl_xor(v, 4, 32));
  v = fmaxf(v, __shfl_xor(v, 8, 32));
  return v;
}
static __device__ inline float redsum16(float v) {
  v += __shfl_xor(v, 1, 32);
  v += __shfl_xor(v, 2, 32);
  v += __shfl_xor(v, 4, 32);
  v += __shfl_xor(v, 8, 32);
  return v;
}

// ---------------------------------------------------------------------------
// Kernel 1: fp32 -> bf16 conversion, 4 elements/thread.
// ---------------------------------------------------------------------------
__global__ __launch_bounds__(256) void cvt_f32_bf16(
    const float* __restrict__ s, __bf16* __restrict__ d, int n4) {
  int i = blockIdx.x * 256 + threadIdx.x;
  if (i < n4) {
    const float4 f = reinterpret_cast<const float4*>(s)[i];
    v4bf o;
    o[0] = (__bf16)f.x; o[1] = (__bf16)f.y;
    o[2] = (__bf16)f.z; o[3] = (__bf16)f.w;
    reinterpret_cast<v4bf*>(d)[i] = o;
  }
}

// ---------------------------------------------------------------------------
// Kernel 2: QKV projection + RoPE. One wave computes a 32-token x 64-dim
// (one full head) tile of one of {Q,K,V}: 8 independent WMMA accumulator
// chains, B fragments amortized over two A rows (12 loads / 8 WMMA per
// k-step). nn.Linear is x @ W.T, so W's native [out,in] row-major layout
// feeds the B-fragment loader directly.
// Q,K stored bf16 [B,H,S,64]; V stored transposed bf16 [B,H,64,S].
// ---------------------------------------------------------------------------
__global__ __launch_bounds__(128) void qkv_rope_wmma(
    const __bf16* __restrict__ xb, const __bf16* __restrict__ wq,
    const __bf16* __restrict__ wk, const __bf16* __restrict__ wv,
    __bf16* __restrict__ Q, __bf16* __restrict__ K, __bf16* __restrict__ Vt) {
  const int wave = blockIdx.x * 4 + (threadIdx.x >> 5);
  const int lane = threadIdx.x & 31;
  const int half = lane >> 4;
  const int col = lane & 15;
  const int mt = wave / 48;        // 32-token tile index (0..127)
  const int rem = wave % 48;
  const int which = rem >> 4;      // 0=Q 1=K 2=V
  const int h = rem & 15;          // head
  const __bf16* W = (which == 0) ? wq : (which == 1) ? wk : wv;
  const int tBase = mt * 32;

  v8f acc[2][4];
#pragma unroll
  for (int m = 0; m < 2; ++m)
#pragma unroll
    for (int n = 0; n < 4; ++n) acc[m][n] = vzero8();

  const int arow0 = tBase + col;
  const int arow1 = tBase + 16 + col;
  for (int k0 = 0; k0 < D_MODEL; k0 += 32) {
    if (k0 + 32 < D_MODEL)
      __builtin_prefetch(W + (size_t)(h * HEAD_DIM + col) * D_MODEL + k0 + 32,
                         0, 1);
    v16bf a0 = load_frag(xb, D_MODEL, arow0, k0, half);
    v16bf a1 = load_frag(xb, D_MODEL, arow1, k0, half);
#pragma unroll
    for (int n = 0; n < 4; ++n) {
      v16bf b = load_frag(W, D_MODEL, h * HEAD_DIM + n * 16 + col, k0, half);
      acc[0][n] = wmma_bf16(a0, b, acc[0][n]);
      acc[1][n] = wmma_bf16(a1, b, acc[1][n]);
    }
  }

  const int bb = tBase >> 11;        // batch (tiles never cross: 2048%32==0)
  const int sTile = tBase & (SEQ - 1);

  if (which < 2) {
    __bf16* out = ((which == 0) ? Q : K) +
                  (size_t)(bb * N_HEADS + h) * SEQ * HEAD_DIM;
#pragma unroll
    for (int m = 0; m < 2; ++m) {
      const int sBase = sTile + m * 16;
      // RoPE in accumulator registers: dim d pairs with d+32, i.e. acc tile t
      // pairs with tile t+2 at identical lane/VGPR. cos/sin identical for the
      // pair since inv_freq depends on d%32. ln(10000) = 9.21034037...
#pragma unroll
      for (int t = 0; t < 2; ++t) {
        const int d = t * 16 + col;  // 0..31
        const float invf = __expf(-(float)d * (9.21034037f / 32.0f));
#pragma unroll
        for (int r = 0; r < 8; ++r) {
          const float pos = (float)(sBase + r + 8 * half);
          const float ang = pos * invf;
          const float sn = __sinf(ang), cs = __cosf(ang);
          const float q1 = acc[m][t][r], q2 = acc[m][t + 2][r];
          acc[m][t][r] = q1 * cs - q2 * sn;
          acc[m][t + 2][r] = q2 * cs + q1 * sn;
        }
      }
#pragma unroll
      for (int n = 0; n < 4; ++n)
#pragma unroll
        for (int r = 0; r < 8; ++r)
          out[(size_t)(sBase + r + 8 * half) * HEAD_DIM + n * 16 + col] =
              (__bf16)acc[m][n][r];
    }
  } else {
    // V transposed: element (token, dim) -> Vt[d][s]; the 8 VGPRs of a
    // fragment column are 8 consecutive tokens -> one contiguous 16B store.
    __bf16* out = Vt + (size_t)(bb * N_HEADS + h) * HEAD_DIM * SEQ;
#pragma unroll
    for (int m = 0; m < 2; ++m) {
      const int sBase = sTile + m * 16;
#pragma unroll
      for (int n = 0; n < 4; ++n) {
        v8bf v;
#pragma unroll
        for (int r = 0; r < 8; ++r) v[r] = (__bf16)acc[m][n][r];
        *reinterpret_cast<v8bf*>(out + (size_t)(n * 16 + col) * SEQ + sBase +
                                 8 * half) = v;
      }
    }
  }
}

// ---------------------------------------------------------------------------
// Kernel 3: causal flash attention. One wave owns (b, h, 16-query tile) and
// streams 32-key blocks with online softmax. The 4 waves of a block cover 4
// consecutive q-tiles of the same (b,h), so K/V fragment loads hit L0/L2.
// P is re-laid-out C-frag -> A-frag through a private 1KB LDS buffer
// (per-wave DS ops are in-order, no barrier needed).
// ---------------------------------------------------------------------------
__global__ __launch_bounds__(128) void flash_attn_wmma(
    const __bf16* __restrict__ Q, const __bf16* __restrict__ K,
    const __bf16* __restrict__ Vt, __bf16* __restrict__ ctx) {
  __shared__ __bf16 pbuf[4][16 * 32];
  const int wslot = threadIdx.x >> 5;
  const int wave = blockIdx.x * 4 + wslot;
  const int lane = threadIdx.x & 31;
  const int half = lane >> 4;
  const int col = lane & 15;
  const int qt = wave & 127;  // query tile within (b,h)
  const int bh = wave >> 7;   // 0..31
  const __bf16* q = Q + (size_t)bh * SEQ * HEAD_DIM;
  const __bf16* k = K + (size_t)bh * SEQ * HEAD_DIM;
  const __bf16* vt = Vt + (size_t)bh * HEAD_DIM * SEQ;
  const int qBase = qt * 16;

  // Q tile fragments reused across all key blocks (k0 = 0, 32 over Dh=64).
  v16bf qf0 = load_frag(q, HEAD_DIM, qBase + col, 0, half);
  v16bf qf1 = load_frag(q, HEAD_DIM, qBase + col, 32, half);

  v8f ctxf[4];
#pragma unroll
  for (int n = 0; n < 4; ++n) ctxf[n] = vzero8();
  float m_run[8], l_run[8];
#pragma unroll
  for (int r = 0; r < 8; ++r) {
    m_run[r] = -3.0e38f;
    l_run[r] = 0.f;
  }

  const int kBlocks = (qBase + 16 + 31) >> 5;  // causal bound
  __bf16* pb = pbuf[wslot];

  for (int kb = 0; kb < kBlocks; ++kb) {
    const int kBase = kb * 32;
    // scores: 16 queries x 32 keys = two 16x16 C tiles, 2 WMMA each over Dh
    v8f sf[2];
    sf[0] = vzero8();
    sf[1] = vzero8();
#pragma unroll
    for (int f = 0; f < 2; ++f) {
      v16bf b0 = load_frag(k, HEAD_DIM, kBase + f * 16 + col, 0, half);
      sf[f] = wmma_bf16(qf0, b0, sf[f]);
      v16bf b1 = load_frag(k, HEAD_DIM, kBase + f * 16 + col, 32, half);
      sf[f] = wmma_bf16(qf1, b1, sf[f]);
    }
    // scale (1/sqrt(64)) + causal mask
#pragma unroll
    for (int f = 0; f < 2; ++f) {
      const int kk = kBase + f * 16 + col;
#pragma unroll
      for (int r = 0; r < 8; ++r) {
        const int qq = qBase + r + 8 * half;
        const float sv = sf[f][r] * 0.125f;
        sf[f][r] = (kk <= qq) ? sv : -1.0e30f;
      }
    }
    // online softmax: row stats live replicated across each 16-lane half
    float alpha[8];
#pragma unroll
    for (int r = 0; r < 8; ++r) {
      const float mx = redmax16(fmaxf(sf[0][r], sf[1][r]));
      const float m_new = fmaxf(m_run[r], mx);
      alpha[r] = __expf(m_run[r] - m_new);
      m_run[r] = m_new;
    }
    float rowsum[8];
#pragma unroll
    for (int r = 0; r < 8; ++r) rowsum[r] = 0.f;
#pragma unroll
    for (int f = 0; f < 2; ++f)
#pragma unroll
      for (int r = 0; r < 8; ++r) {
        const float p = __expf(sf[f][r] - m_run[r]);
        sf[f][r] = p;
        rowsum[r] += p;
      }
#pragma unroll
    for (int r = 0; r < 8; ++r)
      l_run[r] = l_run[r] * alpha[r] + redsum16(rowsum[r]);
#pragma unroll
    for (int n = 0; n < 4; ++n)
#pragma unroll
      for (int r = 0; r < 8; ++r) ctxf[n][r] *= alpha[r];

    // P: C-fragment layout -> LDS -> A-fragment layout (16x32 bf16)
#pragma unroll
    for (int f = 0; f < 2; ++f)
#pragma unroll
      for (int r = 0; r < 8; ++r)
        pb[(r + 8 * half) * 32 + f * 16 + col] = (__bf16)sf[f][r];
    v8bf p0 = *reinterpret_cast<const v8bf*>(pb + col * 32 + 8 * half);
    v8bf p1 = *reinterpret_cast<const v8bf*>(pb + col * 32 + 16 + 8 * half);
    v16bf pf = cat16(p0, p1);

    // ctx += P @ V  (B fragments directly from transposed V rows)
#pragma unroll
    for (int n = 0; n < 4; ++n) {
      v16bf vf = load_frag(vt, SEQ, n * 16 + col, kBase, half);
      ctxf[n] = wmma_bf16(pf, vf, ctxf[n]);
    }
  }

  // normalize and store context bf16 [B*S, D_MODEL] for the final projection
  const int bb = bh >> 4, h = bh & 15;
  __bf16* out = ctx + (size_t)(bb * SEQ + qBase) * D_MODEL + h * HEAD_DIM;
#pragma unroll
  for (int r = 0; r < 8; ++r) {
    const float inv = 1.0f / l_run[r];
#pragma unroll
    for (int n = 0; n < 4; ++n)
      out[(size_t)(r + 8 * half) * D_MODEL + n * 16 + col] =
          (__bf16)(ctxf[n][r] * inv);
  }
}

// ---------------------------------------------------------------------------
// Kernel 4: output projection ctx @ Wo.T -> fp32 out. 32-token x 64-col tile
// per wave (same intensity argument as kernel 2).
// ---------------------------------------------------------------------------
__global__ __launch_bounds__(128) void out_proj_wmma(
    const __bf16* __restrict__ ctx, const __bf16* __restrict__ wo,
    float* __restrict__ out) {
  const int wave = blockIdx.x * 4 + (threadIdx.x >> 5);
  const int lane = threadIdx.x & 31;
  const int half = lane >> 4;
  const int col = lane & 15;
  const int mt = wave >> 4;    // 32-token tile 0..127
  const int nt = wave & 15;    // 64-wide col tile 0..15
  const int tBase = mt * 32, nBase = nt * 64;

  v8f acc[2][4];
#pragma unroll
  for (int m = 0; m < 2; ++m)
#pragma unroll
    for (int n = 0; n < 4; ++n) acc[m][n] = vzero8();

  for (int k0 = 0; k0 < D_MODEL; k0 += 32) {
    if (k0 + 32 < D_MODEL)
      __builtin_prefetch(wo + (size_t)(nBase + col) * D_MODEL + k0 + 32, 0, 1);
    v16bf a0 = load_frag(ctx, D_MODEL, tBase + col, k0, half);
    v16bf a1 = load_frag(ctx, D_MODEL, tBase + 16 + col, k0, half);
#pragma unroll
    for (int n = 0; n < 4; ++n) {
      v16bf b = load_frag(wo, D_MODEL, nBase + n * 16 + col, k0, half);
      acc[0][n] = wmma_bf16(a0, b, acc[0][n]);
      acc[1][n] = wmma_bf16(a1, b, acc[1][n]);
    }
  }
#pragma unroll
  for (int m = 0; m < 2; ++m)
#pragma unroll
    for (int n = 0; n < 4; ++n)
#pragma unroll
      for (int r = 0; r < 8; ++r)
        out[(size_t)(tBase + m * 16 + r + 8 * half) * D_MODEL + nBase +
            n * 16 + col] = acc[m][n][r];
}

// ---------------------------------------------------------------------------
// Host launcher. Inputs: x, attn_mask(ignored: causal known), Wq, Wk, Wv, Wo.
// Workspace layout (bytes): xb@0 (8Mi), wq@8Mi, wk@10Mi, wv@12Mi, wo@14Mi,
// Q@16Mi (8Mi), K@24Mi (8Mi), Vt@32Mi (8Mi), ctx@40Mi (8Mi)  => 48 MiB.
// ---------------------------------------------------------------------------
extern "C" void kernel_launch(void* const* d_in, const int* in_sizes, int n_in,
                              void* d_out, int out_size, void* d_ws,
                              size_t ws_size, hipStream_t stream) {
  (void)in_sizes; (void)n_in; (void)out_size; (void)ws_size;
  const float* x  = (const float*)d_in[0];
  const float* Wq = (const float*)d_in[2];
  const float* Wk = (const float*)d_in[3];
  const float* Wv = (const float*)d_in[4];
  const float* Wo = (const float*)d_in[5];
  float* out = (float*)d_out;

  char* ws = (char*)d_ws;
  const size_t MiB = (size_t)1 << 20;
  __bf16* xb  = (__bf16*)(ws + 0 * MiB);
  __bf16* wqb = (__bf16*)(ws + 8 * MiB);
  __bf16* wkb = (__bf16*)(ws + 10 * MiB);
  __bf16* wvb = (__bf16*)(ws + 12 * MiB);
  __bf16* wob = (__bf16*)(ws + 14 * MiB);
  __bf16* Qb  = (__bf16*)(ws + 16 * MiB);
  __bf16* Kb  = (__bf16*)(ws + 24 * MiB);
  __bf16* Vtb = (__bf16*)(ws + 32 * MiB);
  __bf16* ctx = (__bf16*)(ws + 40 * MiB);

  const int nx4 = NTOK * D_MODEL / 4;      // 1,048,576
  const int nw4 = D_MODEL * D_MODEL / 4;   // 262,144
  cvt_f32_bf16<<<(nx4 + 255) / 256, 256, 0, stream>>>(x, xb, nx4);
  cvt_f32_bf16<<<(nw4 + 255) / 256, 256, 0, stream>>>(Wq, wqb, nw4);
  cvt_f32_bf16<<<(nw4 + 255) / 256, 256, 0, stream>>>(Wk, wkb, nw4);
  cvt_f32_bf16<<<(nw4 + 255) / 256, 256, 0, stream>>>(Wv, wvb, nw4);
  cvt_f32_bf16<<<(nw4 + 255) / 256, 256, 0, stream>>>(Wo, wob, nw4);

  // (NTOK/32) tiles * 16 heads * 3 mats = 6144 waves / 4 per block
  qkv_rope_wmma<<<1536, 128, 0, stream>>>(xb, wqb, wkb, wvb, Qb, Kb, Vtb);
  // B*H * S/16 = 4096 waves / 4 per block
  flash_attn_wmma<<<1024, 128, 0, stream>>>(Qb, Kb, Vtb, ctx);
  // (NTOK/32) * (D_MODEL/64) = 2048 waves / 4 per block
  out_proj_wmma<<<512, 128, 0, stream>>>(ctx, wob, out);
}
